// FFJORDDensity_48146583388237
// MI455X (gfx1250) — matrix operand order repaired and checked by reading, not compiled, not run on hardware
//
#include <hip/hip_runtime.h>

// ---------------------------------------------------------------------------
// FFJORD density on MI455X (gfx1250): persistent per-workgroup RK4 integration,
// all 6 GEMMs per dynamics eval on v_wmma_f32_16x16x32_bf16, weights streamed
// from L2-resident bf16 copies (+ transposes) built by a prep kernel.
//
// Round-2 fix: weight base pointers are laundered through an asm volatile
// no-op per GEMM call so LICM cannot hoist the (loop-invariant) weight
// fragment loads out of the step/stage loops -- round 1 showed the compiler
// hoisting them, blowing past 256 VGPRs (s_set_vgpr_msb) and feeding every
// WMMA from scratch spill reloads.
// ---------------------------------------------------------------------------

#define BDIM 128       // data dim
#define HID  512       // hidden dim
#define MT   32        // batch rows per workgroup
#define NTHREADS 256   // 8 wave32s

typedef __bf16 v16bf __attribute__((ext_vector_type(16)));
typedef float  v8f   __attribute__((ext_vector_type(8)));

union FragBF { int4 q[2]; v16bf v; };

__device__ __forceinline__ float tanh_fast(float x) {
  // tanh(x) = 1 - 2/(exp(2x)+1); v_exp_f32 + v_rcp_f32 path, correct limits.
  return 1.0f - 2.0f / (__expf(2.0f * x) + 1.0f);
}

// Opaque pointer laundering: defeats LICM/CSE of weight loads across the
// integration loops while keeping the base uniform (SGPR pair).
__device__ __forceinline__ const __bf16* opaque_ptr(const __bf16* p) {
  asm volatile("" : "+s"(p));
  return p;
}

// ---------------------------------------------------------------------------
// One wave computes NT 16x16 tiles of D = A[32,K] x B[K,N] for its row block
// rb (0/1) and column origin n0.  A is bf16 in LDS (row-major, stride K).
// Bt is the GLOBAL bf16 transposed weight [N,K] row-major (L2-resident).
// A-frag (this code's k order): lane l -> row (l&15), contiguous 16 K at k=(l>>4)*16.
// B-frag (same k order as A): lane l -> col (l&15), contiguous 16 K at k=(l>>4)*16.
// ---------------------------------------------------------------------------
template<int K, int NT>
__device__ __forceinline__ void wmma_gemm(const __bf16* __restrict__ Alds,
                                          const __bf16* __restrict__ Bt_in,
                                          int rb, int n0, v8f* acc) {
  const __bf16* Bt = opaque_ptr(Bt_in);   // fresh value per call site/iteration
  const int lane = threadIdx.x & 31;
  const int l15  = lane & 15;
  const int hof  = lane >> 4;
  const __bf16* Abase = Alds + (rb * 16 + l15) * K;
  const __bf16* Bbase0 = Bt + (n0 + l15) * K + hof * 16;
#pragma unroll 1
  for (int kc = 0; kc < K / 32; ++kc) {
    FragBF a;
    const int ak = kc * 32 + hof * 16;
    a.q[0] = *reinterpret_cast<const int4*>(Abase + ak);
    a.q[1] = *reinterpret_cast<const int4*>(Abase + ak + 8);
#pragma unroll
    for (int t = 0; t < NT; ++t) {
      const __bf16* Bbase = Bbase0 + t * 16 * K + kc * 32;
      FragBF b;
      b.q[0] = *reinterpret_cast<const int4*>(Bbase);
      b.q[1] = *reinterpret_cast<const int4*>(Bbase + 8);
      acc[t] = __builtin_amdgcn_wmma_f32_16x16x32_bf16(
          false, a.v, false, b.v, (short)0, acc[t], false, false);
    }
  }
}

// C/D layout: lane l, vgpr j -> m = rb*16 + j + 8*(l>>4), n = n0 + 16*t + (l&15)

template<int NT>
__device__ __forceinline__ void epi_tanh_bf16(const v8f* acc, int rb, int n0,
                                              const float* __restrict__ bias,
                                              const float* __restrict__ trow,
                                              float ts, __bf16* __restrict__ out,
                                              int ns) {
  const int lane = threadIdx.x & 31;
  const int l15 = lane & 15, hof = lane >> 4;
#pragma unroll
  for (int t = 0; t < NT; ++t) {
    const int n = n0 + t * 16 + l15;
    float bv = bias[n];
    if (trow) bv += ts * trow[n];
#pragma unroll
    for (int j = 0; j < 8; ++j) {
      const int m = rb * 16 + j + 8 * hof;
      out[m * ns + n] = (__bf16)tanh_fast(acc[t][j] + bv);
    }
  }
}

template<int NT>
__device__ __forceinline__ void epi_bias_f32(const v8f* acc, int rb, int n0,
                                             const float* __restrict__ bias,
                                             float* __restrict__ out, int ns) {
  const int lane = threadIdx.x & 31;
  const int l15 = lane & 15, hof = lane >> 4;
#pragma unroll
  for (int t = 0; t < NT; ++t) {
    const int n = n0 + t * 16 + l15;
    const float bv = bias[n];
#pragma unroll
    for (int j = 0; j < 8; ++j) {
      const int m = rb * 16 + j + 8 * hof;
      out[m * ns + n] = acc[t][j] + bv;
    }
  }
}

// g' = g * (1 - h^2), reading/overwriting the same LDS element (in-place safe).
template<int NT>
__device__ __forceinline__ void epi_dtanh(const v8f* acc, int rb, int n0,
                                          __bf16* __restrict__ hbuf, int ns) {
  const int lane = threadIdx.x & 31;
  const int l15 = lane & 15, hof = lane >> 4;
#pragma unroll
  for (int t = 0; t < NT; ++t) {
    const int n = n0 + t * 16 + l15;
#pragma unroll
    for (int j = 0; j < 8; ++j) {
      const int m = rb * 16 + j + 8 * hof;
      const float h = (float)hbuf[m * ns + n];
      hbuf[m * ns + n] = (__bf16)(acc[t][j] * (1.0f - h * h));
    }
  }
}

// div[m] += sum_n vJ[m,n] * v[m,n]
template<int NT>
__device__ __forceinline__ void epi_div(const v8f* acc, int rb, int n0,
                                        const float* __restrict__ vbuf,
                                        float* __restrict__ divb) {
  const int lane = threadIdx.x & 31;
  const int l15 = lane & 15, hof = lane >> 4;
#pragma unroll
  for (int t = 0; t < NT; ++t) {
    const int n = n0 + t * 16 + l15;
#pragma unroll
    for (int j = 0; j < 8; ++j) {
      const int m = rb * 16 + j + 8 * hof;
      atomicAdd(&divb[m], acc[t][j] * vbuf[m * BDIM + n]);
    }
  }
}

// ---------------------------------------------------------------------------
// Prep: bf16 copies of W1[0:128,:], W2, W3 plus transposes into workspace.
// Offsets (bf16 elements): W1p 0, W1pT 65536, W2 131072, W2T 393216,
//                          W3 655360, W3T 720896.  Total 1.5 MB.
// ---------------------------------------------------------------------------
__global__ void prep_weights(const float* __restrict__ W1,
                             const float* __restrict__ W2,
                             const float* __restrict__ W3,
                             unsigned short* __restrict__ wraw) {
  __bf16* wbuf = reinterpret_cast<__bf16*>(wraw);
  __bf16* W1p  = wbuf;
  __bf16* W1pT = wbuf + 65536;
  __bf16* W2b  = wbuf + 131072;
  __bf16* W2Tb = wbuf + 393216;
  __bf16* W3b  = wbuf + 655360;
  __bf16* W3Tb = wbuf + 720896;
  const int i = blockIdx.x * blockDim.x + threadIdx.x;
  if (i < 128 * 512) {  // W1 without the t-row
    const int r = i >> 9, c = i & 511;
    const __bf16 w = (__bf16)W1[i];
    W1p[i] = w; W1pT[c * 128 + r] = w;
  }
  if (i < 512 * 512) {
    const int r = i >> 9, c = i & 511;
    const __bf16 w = (__bf16)W2[i];
    W2b[i] = w; W2Tb[c * 512 + r] = w;
  }
  if (i < 512 * 128) {
    const int r = i >> 7, c = i & 127;
    const __bf16 w = (__bf16)W3[i];
    W3b[i] = w; W3Tb[c * 512 + r] = w;
  }
}

// ---------------------------------------------------------------------------
// Main: each workgroup owns MT=32 batch rows, integrates all 32 RK4 steps.
// ---------------------------------------------------------------------------
__global__ __launch_bounds__(NTHREADS)
void ffjord_kernel(const float* __restrict__ x, const float* __restrict__ v,
                   const float* __restrict__ b1, const float* __restrict__ b2,
                   const float* __restrict__ b3, const float* __restrict__ W1full,
                   const unsigned short* __restrict__ wraw,
                   float* __restrict__ out) {
  __shared__ __attribute__((aligned(16))) __bf16 abuf[MT * HID];   // h1 / g1'
  __shared__ __attribute__((aligned(16))) __bf16 bbuf[MT * HID];   // h2 / g2'
  __shared__ __attribute__((aligned(16))) __bf16 zsbuf[MT * BDIM]; // stage z, bf16
  __shared__ __attribute__((aligned(16))) __bf16 vbf[MT * BDIM];   // probe v, bf16
  __shared__ float zbuf[MT * BDIM];
  __shared__ float vbuf[MT * BDIM];
  __shared__ float fbuf[MT * BDIM];
  __shared__ float kacc[MT * BDIM];
  __shared__ float divb[MT];
  __shared__ float kdiv[MT];
  __shared__ float logp[MT];

  const __bf16* wbuf = reinterpret_cast<const __bf16*>(wraw);
  const __bf16* W1p  = wbuf;            // [128,512] = Bt for vJ gemm
  const __bf16* W1pT = wbuf + 65536;    // [512,128] = Bt for a1 gemm
  const __bf16* W2b  = wbuf + 131072;   // [512,512] = Bt for g1 gemm
  const __bf16* W2Tb = wbuf + 393216;   // [512,512] = Bt for a2 gemm
  const __bf16* W3b  = wbuf + 655360;   // [512,128] = Bt for g2 gemm
  const __bf16* W3Tb = wbuf + 720896;   // [128,512] = Bt for f  gemm
  const float* w1row = W1full + 128 * HID;  // time column of W1

  const int tid  = threadIdx.x;
  const int wave = tid >> 5;
  const int rb   = wave & 1;            // row block (0/1)
  const int n0w  = (wave >> 1) * 128;   // N=512 gemms: 8 tiles per wave
  const int n0s  = (wave >> 1) * 32;    // N=128 gemms: 2 tiles per wave
  const long row0 = (long)blockIdx.x * MT;

  for (int i = tid; i < MT * BDIM; i += NTHREADS) {
    const float zv = x[row0 * BDIM + i];
    const float vv = v[row0 * BDIM + i];
    zbuf[i] = zv; vbuf[i] = vv;
    zsbuf[i] = (__bf16)zv; vbf[i] = (__bf16)vv;
    kacc[i] = 0.0f;
  }
  if (tid < MT) { divb[tid] = 0.0f; kdiv[tid] = 0.0f; logp[tid] = 0.0f; }
  __syncthreads();

  const float dt = 1.0f / 32.0f;
  const v8f vzero = {0.f, 0.f, 0.f, 0.f, 0.f, 0.f, 0.f, 0.f};

#pragma unroll 1
  for (int step = 0; step < 32; ++step) {
    const float t0 = (float)step * dt;
#pragma unroll 1
    for (int s = 0; s < 4; ++s) {
      const float cs = (s == 0) ? 0.0f : ((s == 3) ? 1.0f : 0.5f);
      const float ws = (s == 1 || s == 2) ? 2.0f : 1.0f;
      const float ts = t0 + cs * dt;

      // ---- forward: h1 = tanh(zs @ W1p + b1 + ts*W1[t-row]) ----
      {
        v8f acc[8]; for (int t = 0; t < 8; ++t) acc[t] = vzero;
        wmma_gemm<BDIM, 8>(zsbuf, W1pT, rb, n0w, acc);
        epi_tanh_bf16<8>(acc, rb, n0w, b1, w1row, ts, abuf, HID);
      }
      __syncthreads();
      // ---- h2 = tanh(h1 @ W2 + b2) ----
      {
        v8f acc[8]; for (int t = 0; t < 8; ++t) acc[t] = vzero;
        wmma_gemm<HID, 8>(abuf, W2Tb, rb, n0w, acc);
        epi_tanh_bf16<8>(acc, rb, n0w, b2, nullptr, 0.0f, bbuf, HID);
      }
      __syncthreads();
      // ---- f = h2 @ W3 + b3 ----
      {
        v8f acc[2]; acc[0] = vzero; acc[1] = vzero;
        wmma_gemm<HID, 2>(bbuf, W3Tb, rb, n0s, acc);
        epi_bias_f32<2>(acc, rb, n0s, b3, fbuf, BDIM);
      }
      __syncthreads();  // all reads of h2 done before in-place g2' overwrite
      // ---- backward: g2' = (v @ W3^T) * (1 - h2^2)  (overwrites bbuf) ----
      {
        v8f acc[8]; for (int t = 0; t < 8; ++t) acc[t] = vzero;
        wmma_gemm<BDIM, 8>(vbf, W3b, rb, n0w, acc);
        epi_dtanh<8>(acc, rb, n0w, bbuf, HID);
      }
      __syncthreads();
      // ---- g1' = (g2' @ W2^T) * (1 - h1^2)  (overwrites abuf) ----
      {
        v8f acc[8]; for (int t = 0; t < 8; ++t) acc[t] = vzero;
        wmma_gemm<HID, 8>(bbuf, W2b, rb, n0w, acc);
        epi_dtanh<8>(acc, rb, n0w, abuf, HID);
      }
      __syncthreads();
      // ---- div[m] = sum_n (g1' @ W1p^T)[m,n] * v[m,n] ----
      {
        v8f acc[2]; acc[0] = vzero; acc[1] = vzero;
        wmma_gemm<HID, 2>(abuf, W1p, rb, n0s, acc);
        epi_div<2>(acc, rb, n0s, vbuf, divb);
      }
      __syncthreads();

      // ---- RK4 stage accumulation ----
      const float cnext = (s <= 1) ? 0.5f : 1.0f;
      for (int i = tid; i < MT * BDIM; i += NTHREADS) {
        const float fv = fbuf[i];
        kacc[i] += ws * fv;
        if (s < 3) zsbuf[i] = (__bf16)(zbuf[i] + cnext * dt * fv);
      }
      if (tid < MT) { kdiv[tid] += ws * divb[tid]; divb[tid] = 0.0f; }
      __syncthreads();
    }
    // ---- z, logp update ----
    for (int i = tid; i < MT * BDIM; i += NTHREADS) {
      const float zn = zbuf[i] + (dt / 6.0f) * kacc[i];
      zbuf[i] = zn; kacc[i] = 0.0f;
      zsbuf[i] = (__bf16)zn;
    }
    if (tid < MT) { logp[tid] -= (dt / 6.0f) * kdiv[tid]; kdiv[tid] = 0.0f; }
    __syncthreads();
  }

  // ---- prior log-density + accumulated log-det ----
  for (int i = tid; i < MT * BDIM; i += NTHREADS) {
    const float zv = zbuf[i];
    atomicAdd(&divb[i >> 7], zv * zv);  // row = i / 128
  }
  __syncthreads();
  if (tid < MT) {
    const float log2pi = 1.8378770664093453f;
    out[row0 + tid] = -0.5f * divb[tid] - 0.5f * (float)BDIM * log2pi + logp[tid];
  }
}

// ---------------------------------------------------------------------------
extern "C" void kernel_launch(void* const* d_in, const int* in_sizes, int n_in,
                              void* d_out, int out_size, void* d_ws, size_t ws_size,
                              hipStream_t stream) {
  const float* x  = (const float*)d_in[0];
  const float* v  = (const float*)d_in[1];
  const float* W1 = (const float*)d_in[2];
  const float* b1 = (const float*)d_in[3];
  const float* W2 = (const float*)d_in[4];
  const float* b2 = (const float*)d_in[5];
  const float* W3 = (const float*)d_in[6];
  const float* b3 = (const float*)d_in[7];
  float* out = (float*)d_out;
  unsigned short* wbuf = (unsigned short*)d_ws;  // 786432 bf16 = 1.5 MB

  const int B = in_sizes[0] / BDIM;  // 65536

  prep_weights<<<(512 * 512 + 255) / 256, 256, 0, stream>>>(W1, W2, W3, wbuf);
  ffjord_kernel<<<B / MT, NTHREADS, 0, stream>>>(x, v, b1, b2, b3, W1, wbuf, out);
}
